// QuantizedLookupClassifier_14113262535113
// MI455X (gfx1250) — matrix operand order repaired and checked by reading, not compiled, not run on hardware
//
#include <hip/hip_runtime.h>
#include <hip/hip_bf16.h>

// ---------------------------------------------------------------------------
// Problem constants (from the reference)
// ---------------------------------------------------------------------------
#define NUM_CODEBOOKS 8
#define CODEBOOK_SIZE 4096
#define SUBVEC_DIM 128
#define NUM_CLASSES 1000
#define BATCH 8192
#define INPUT_DIM (NUM_CODEBOOKS * SUBVEC_DIM)

typedef __attribute__((ext_vector_type(16))) __bf16 v16bf;
typedef __attribute__((ext_vector_type(8)))  float  v8f;

// Round-to-nearest-even f32 -> bf16
__device__ __forceinline__ __bf16 f2bf(float f) {
    unsigned u = __builtin_bit_cast(unsigned, f);
    unsigned r = u + 0x7FFFu + ((u >> 16) & 1u);
    unsigned short s = (unsigned short)(r >> 16);
    return __builtin_bit_cast(__bf16, s);
}

// ---------------------------------------------------------------------------
// Kernel 1: convert codebooks f32 -> bf16 and compute ||c||^2 of the rounded
// values. One wave per code row (128 dims); 8 waves per 256-thread block.
// ---------------------------------------------------------------------------
__global__ __launch_bounds__(256)
void convert_codebooks(const float* __restrict__ cb,
                       __bf16* __restrict__ cbb,
                       float* __restrict__ csq) {
    const int wave = threadIdx.x >> 5;
    const int lane = threadIdx.x & 31;
    const int row  = blockIdx.x * 8 + wave;     // row in [0, 8*4096)
    const float* src = cb  + (size_t)row * SUBVEC_DIM;
    __bf16*      dst = cbb + (size_t)row * SUBVEC_DIM;
    float ssum = 0.0f;
    #pragma unroll
    for (int i = lane; i < SUBVEC_DIM; i += 32) {
        __bf16 b = f2bf(src[i]);
        dst[i] = b;
        float fb = (float)b;
        ssum += fb * fb;
    }
    #pragma unroll
    for (int m = 16; m >= 1; m >>= 1) ssum += __shfl_xor(ssum, m, 32);
    if (lane == 0) csq[row] = ssum;
}

// ---------------------------------------------------------------------------
// Kernel 2: fused distance GEMM + argmin, software-pipelined.
// Grid: (BATCH/256, NUM_CODEBOOKS), block = 256 threads = 8 waves.
// Wave w owns rows [tile + 32w, tile + 32w + 32) as two 16-row fragments and
// streams all 4096 codes in 16-wide chunks via chained
// v_wmma_f32_16x16x32_bf16 (K = 128 = 4 x 32). B fragments + ||c||^2 are
// double-buffered (loop unrolled by 2 with alternating register sets) so the
// global loads for chunk n+1 overlap the WMMAs of chunk n.
// ---------------------------------------------------------------------------
#define MTILE 256

// Load one 16-code B chunk (4 K-fragments, ISA 16-bit B 32x16 layout) + csq.
// lane l: N = l%16; VGPR v: K = 32q + (v>=4 ? 16:0) + (l>=16 ? 8:0) + 2*(v&3) + {0,1}
__device__ __forceinline__ void load_bchunk(const unsigned* __restrict__ cbu,
                                            const float* __restrict__ csqk,
                                            int chunk, int lmod, int lhi,
                                            v16bf Bf[4], float& cs) {
    const int N = chunk + lmod;
    cs = csqk[N];
    const unsigned* crow = cbu + (size_t)N * (SUBVEC_DIM / 2);  // 64 dwords/row
    #pragma unroll
    for (int q = 0; q < 4; ++q) {
        #pragma unroll
        for (int v = 0; v < 8; ++v) {
            const int K = 32 * q + ((v & 4) ? 16 : 0) + (lhi ? 8 : 0) + 2 * (v & 3);
            const unsigned u = crow[K >> 1];
            Bf[q][2 * v]     = __builtin_bit_cast(__bf16, (unsigned short)(u & 0xFFFFu));
            Bf[q][2 * v + 1] = __builtin_bit_cast(__bf16, (unsigned short)(u >> 16));
        }
    }
}

// K=128 WMMA chain for both row fragments + running argmin update.
// C/D layout: VGPR v -> row (v + 8*(lane>=16)), lane%16 -> N
__device__ __forceinline__ void compute_chunk(const v16bf A[2][4], const v16bf Bf[4],
                                              float cs, int N,
                                              float minv[2][8], int mini[2][8]) {
    #pragma unroll
    for (int f = 0; f < 2; ++f) {
        v8f acc = {};
        #pragma unroll
        for (int q = 0; q < 4; ++q) {
            acc = __builtin_amdgcn_wmma_f32_16x16x32_bf16(
                /*neg_a=*/false, A[f][q], /*neg_b=*/false, Bf[q],
                /*c_mod=*/(short)0, acc, /*reuse_a=*/false, /*reuse_b=*/false);
        }
        #pragma unroll
        for (int v = 0; v < 8; ++v) {
            const float dist = cs - 2.0f * acc[v];
            if (dist < minv[f][v]) { minv[f][v] = dist; mini[f][v] = N; }
        }
    }
}

__global__ __launch_bounds__(256)
void argmin_kernel(const float* __restrict__ x,
                   const __bf16* __restrict__ cbb,
                   const float* __restrict__ csq,
                   int* __restrict__ indices) {
    const int k    = blockIdx.y;
    const int lane = threadIdx.x & 31;
    const int wave = threadIdx.x >> 5;
    const int rowBase = blockIdx.x * MTILE + wave * 32;
    const int lmod = lane & 15;
    const int lhi  = lane >> 4;            // 0: lanes 0-15, 1: lanes 16-31

    // ---- Load A fragments (ISA 16-bit A 16x32 layout), kept in registers ----
    // lane l, VGPR v: K = 32q + (v>=4 ? 16:0) + (l>=16 ? 8:0) + 2*(v&3) + {0,1}
    v16bf A[2][4];
    #pragma unroll
    for (int f = 0; f < 2; ++f) {
        const int R = rowBase + f * 16 + lmod;
        const float* xr = x + (size_t)R * INPUT_DIM + k * SUBVEC_DIM;
        #pragma unroll
        for (int q = 0; q < 4; ++q) {
            #pragma unroll
            for (int v = 0; v < 8; ++v) {
                const int K = 32 * q + ((v & 4) ? 16 : 0) + (lhi ? 8 : 0) + 2 * (v & 3);
                A[f][q][2 * v]     = f2bf(xr[K]);
                A[f][q][2 * v + 1] = f2bf(xr[K + 1]);
            }
        }
    }

    float minv[2][8];
    int   mini[2][8];
    #pragma unroll
    for (int f = 0; f < 2; ++f)
        #pragma unroll
        for (int v = 0; v < 8; ++v) { minv[f][v] = 3.4e38f; mini[f][v] = 0; }

    const float*    csqk = csq + k * CODEBOOK_SIZE;
    const unsigned* cbu  = (const unsigned*)(cbb + (size_t)k * CODEBOOK_SIZE * SUBVEC_DIM);

    // ---- Software-pipelined main loop: 2 chunks per iteration, B0/B1 ping-pong
    v16bf B0[4], B1[4];
    float cs0, cs1;
    load_bchunk(cbu, csqk, 0, lmod, lhi, B0, cs0);

    for (int chunk = 0; chunk < CODEBOOK_SIZE; chunk += 32) {
        // Issue loads for chunk+16 while computing chunk.
        load_bchunk(cbu, csqk, chunk + 16, lmod, lhi, B1, cs1);

        // Prefetch codebook rows two chunks ahead toward the WGP caches
        // (global_prefetch_b8; rows are 256B = 2 cachelines).
        if (chunk + 32 < CODEBOOK_SIZE) {
            const char* pf = (const char*)(cbu +
                (size_t)(chunk + 32 + lmod) * (SUBVEC_DIM / 2));
            __builtin_prefetch(pf, 0, 1);
            __builtin_prefetch(pf + 128, 0, 1);
        }

        compute_chunk(A, B0, cs0, chunk + lmod, minv, mini);

        // Issue loads for chunk+32 while computing chunk+16.
        if (chunk + 32 < CODEBOOK_SIZE)
            load_bchunk(cbu, csqk, chunk + 32, lmod, lhi, B0, cs0);

        compute_chunk(A, B1, cs1, chunk + 16 + lmod, minv, mini);
    }

    // ---- Reduce argmin across the 16 lanes sharing each row ----
    // xor masks 8,4,2,1 stay inside each 16-lane half (distinct row sets).
    #pragma unroll
    for (int f = 0; f < 2; ++f) {
        #pragma unroll
        for (int v = 0; v < 8; ++v) {
            float mv = minv[f][v];
            int   mi = mini[f][v];
            #pragma unroll
            for (int m = 8; m >= 1; m >>= 1) {
                float ov = __shfl_xor(mv, m, 32);
                int   oi = __shfl_xor(mi, m, 32);
                if (ov < mv || (ov == mv && oi < mi)) { mv = ov; mi = oi; }
            }
            if (lmod == 0) {
                const int R = rowBase + f * 16 + lhi * 8 + v;
                indices[R * NUM_CODEBOOKS + k] = mi;
            }
        }
    }
}

// ---------------------------------------------------------------------------
// Kernel 3: gather table rows by argmin index and sum over codebooks.
// One block per batch row; 250 active threads each own one float4 column
// group (1000 f32 = 250 float4, rows are 16B aligned since 4000 % 16 == 0).
// tables (131 MB) is L2-resident on MI455X (192 MB L2).
// ---------------------------------------------------------------------------
__global__ __launch_bounds__(256)
void gather_kernel(const float* __restrict__ tables,
                   const int* __restrict__ indices,
                   float* __restrict__ out) {
    const int b = blockIdx.x;
    const int t = threadIdx.x;
    __shared__ int sidx[NUM_CODEBOOKS];
    if (t < NUM_CODEBOOKS) sidx[t] = indices[b * NUM_CODEBOOKS + t];
    __syncthreads();
    if (t >= NUM_CLASSES / 4) return;
    float4 acc = make_float4(0.0f, 0.0f, 0.0f, 0.0f);
    #pragma unroll
    for (int kk = 0; kk < NUM_CODEBOOKS; ++kk) {
        const float4* row = (const float4*)(tables +
            ((size_t)kk * CODEBOOK_SIZE + sidx[kk]) * NUM_CLASSES);
        float4 vv = row[t];
        acc.x += vv.x; acc.y += vv.y; acc.z += vv.z; acc.w += vv.w;
    }
    ((float4*)(out + (size_t)b * NUM_CLASSES))[t] = acc;
}

// ---------------------------------------------------------------------------
// Launch
// ---------------------------------------------------------------------------
extern "C" void kernel_launch(void* const* d_in, const int* in_sizes, int n_in,
                              void* d_out, int out_size, void* d_ws, size_t ws_size,
                              hipStream_t stream) {
    const float* x         = (const float*)d_in[0];   // [8192, 1024]
    const float* codebooks = (const float*)d_in[1];   // [8, 4096, 128]
    const float* tables    = (const float*)d_in[2];   // [8, 4096, 1000]
    float* out = (float*)d_out;                       // [8192, 1000]

    // Workspace layout (~8.8 MB total):
    char* ws = (char*)d_ws;
    const size_t cbb_bytes = (size_t)NUM_CODEBOOKS * CODEBOOK_SIZE * SUBVEC_DIM * 2; // 8388608
    const size_t csq_bytes = (size_t)NUM_CODEBOOKS * CODEBOOK_SIZE * 4;              // 131072
    __bf16* cbb     = (__bf16*)ws;
    float*  csq     = (float*)(ws + cbb_bytes);
    int*    indices = (int*)(ws + cbb_bytes + csq_bytes);                            // 262144

    // 1) Convert codebooks to bf16 + squared norms (of rounded values).
    convert_codebooks<<<dim3((NUM_CODEBOOKS * CODEBOOK_SIZE) / 8), 256, 0, stream>>>(
        codebooks, cbb, csq);

    // 2) Fused bf16-WMMA distance GEMM + per-row argmin (double-buffered).
    argmin_kernel<<<dim3(BATCH / MTILE, NUM_CODEBOOKS), 256, 0, stream>>>(
        x, cbb, csq, indices);

    // 3) Gather table rows and sum over codebooks.
    gather_kernel<<<dim3(BATCH), 256, 0, stream>>>(tables, indices, out);
}
